// PointNetPPEncoder_80272938762725
// MI455X (gfx1250) — compile-verified, matrix operand order checked by reading, not run on hardware
//
#include <hip/hip_runtime.h>
#include <hip/hip_bf16.h>

typedef __attribute__((ext_vector_type(16))) _Float16 v16h;
typedef __attribute__((ext_vector_type(8)))  _Float16 v8h;
typedef __attribute__((ext_vector_type(8)))  float    v8f;

#define WMMA_F32_F16(a, b, c) \
  __builtin_amdgcn_wmma_f32_16x16x32_f16(false, (a), false, (b), (short)0, (c), false, false)

// ---------------------------------------------------------------------------
// Fragment loader for v_wmma_f32_16x16x32_f16.
// A is 16x32 (MxK) row-major with leading dim `ld` (in halfs).
// Per ISA layout: lane = 16*half + m; VGPR0..3 hold k = half*8 + 0..7,
// VGPR4..7 hold k = 16 + half*8 + 0..7  ->  two contiguous 16B loads/lane.
// The same pattern loads B fragments when B is stored transposed [N][K].
// ---------------------------------------------------------------------------
__device__ __forceinline__ v16h load_frag(const _Float16* __restrict__ base,
                                          int ld, int r0, int kb) {
  const int lane = threadIdx.x & 31;
  const _Float16* p =
      base + (size_t)(r0 + (lane & 15)) * ld + kb + ((lane >> 4) & 1) * 8;
  v8h lo = *reinterpret_cast<const v8h*>(p);
  v8h hi = *reinterpret_cast<const v8h*>(p + 16);
  v16h f;
#pragma unroll
  for (int i = 0; i < 8; ++i) { f[i] = lo[i]; f[i + 8] = hi[i]; }
  return f;
}

// ---------------------------------------------------------------------------
// Weight prep: W f32 [K,N] (row-major, h @ W) -> Wt f16 [N, Kpad], zero-padded.
// ---------------------------------------------------------------------------
__global__ void wcvt_kernel(const float* __restrict__ W, _Float16* __restrict__ Wt,
                            int K, int N, int Kpad) {
  int i = blockIdx.x * blockDim.x + threadIdx.x;
  int total = N * Kpad;
  if (i >= total) return;
  int n = i / Kpad, k = i - n * Kpad;
  Wt[i] = (k < K) ? (_Float16)W[(size_t)k * N + n] : (_Float16)0.f;
}

// ---------------------------------------------------------------------------
// Farthest point sampling. One block (256 threads) per graph.
// Dynamic LDS: md[P] + rv[256] + ri[256].
// ---------------------------------------------------------------------------
#define FPS_T 256
__global__ void fps_kernel(const float* __restrict__ pos, float* __restrict__ posdst,
                           int* __restrict__ idx, int P, int S) {
  extern __shared__ float sm[];
  float* md = sm;
  float* rv = sm + P;
  int*   ri = (int*)(sm + P + FPS_T);
  const int g = blockIdx.x, t = threadIdx.x;
  const float* gp = pos + (size_t)g * P * 3;
  float cx = gp[0], cy = gp[1], cz = gp[2];
  for (int p = t; p < P; p += FPS_T) {
    float dx = gp[p * 3 + 0] - cx, dy = gp[p * 3 + 1] - cy, dz = gp[p * 3 + 2] - cz;
    md[p] = dx * dx + dy * dy + dz * dz;
  }
  if (t == 0) {
    idx[(size_t)g * S] = 0;
    posdst[(size_t)g * S * 3 + 0] = cx;
    posdst[(size_t)g * S * 3 + 1] = cy;
    posdst[(size_t)g * S * 3 + 2] = cz;
  }
  __syncthreads();
  for (int s = 1; s < S; ++s) {
    float bv = -1.f; int bi = 0;
    for (int p = t; p < P; p += FPS_T) {
      float v = md[p];
      if (v > bv) { bv = v; bi = p; }
    }
    rv[t] = bv; ri[t] = bi;
    __syncthreads();
    for (int o = FPS_T / 2; o > 0; o >>= 1) {
      if (t < o && rv[t + o] > rv[t]) { rv[t] = rv[t + o]; ri[t] = ri[t + o]; }
      __syncthreads();
    }
    int nx = ri[0];
    __syncthreads();
    cx = gp[nx * 3 + 0]; cy = gp[nx * 3 + 1]; cz = gp[nx * 3 + 2];
    if (t == 0) {
      idx[(size_t)g * S + s] = nx;
      posdst[((size_t)g * S + s) * 3 + 0] = cx;
      posdst[((size_t)g * S + s) * 3 + 1] = cy;
      posdst[((size_t)g * S + s) * 3 + 2] = cz;
    }
    for (int p = t; p < P; p += FPS_T) {
      float dx = gp[p * 3 + 0] - cx, dy = gp[p * 3 + 1] - cy, dz = gp[p * 3 + 2] - cz;
      float d = dx * dx + dy * dy + dz * dz;
      float o = md[p];
      md[p] = d < o ? d : o;
    }
    __syncthreads();
  }
}

// ---------------------------------------------------------------------------
// 32-nearest-neighbor selection per center, one wave per center.
// Writes nbr[center*32+j] and a 32-bit validity mask (d2 <= r2) per center.
// Dynamic LDS: d2[P].
// ---------------------------------------------------------------------------
__global__ void __launch_bounds__(32) nbr_kernel(const float* __restrict__ pos,
                                                 const float* __restrict__ posdst,
                                                 int* __restrict__ nbr,
                                                 unsigned* __restrict__ maskb,
                                                 int P, int S, float r2) {
  extern __shared__ float d2[];
  const int c = blockIdx.x;         // g*S + s
  const int g = c / S;
  const int lane = threadIdx.x;
  const float* gp = pos + (size_t)g * P * 3;
  const float cx = posdst[(size_t)c * 3 + 0];
  const float cy = posdst[(size_t)c * 3 + 1];
  const float cz = posdst[(size_t)c * 3 + 2];
  for (int p = lane; p < P; p += 32) {
    float dx = gp[p * 3 + 0] - cx, dy = gp[p * 3 + 1] - cy, dz = gp[p * 3 + 2] - cz;
    d2[p] = dx * dx + dy * dy + dz * dz;
  }
  __syncthreads();
  unsigned mb = 0;
  for (int j = 0; j < 32; ++j) {
    float bv = 3.4e38f; int bi = 0;
    for (int p = lane; p < P; p += 32) {
      float v = d2[p];
      if (v < bv) { bv = v; bi = p; }
    }
    for (int o = 16; o > 0; o >>= 1) {
      float ov = __shfl_down(bv, o, 32);
      int   oi = __shfl_down(bi, o, 32);
      if (ov < bv) { bv = ov; bi = oi; }
    }
    bi = __shfl(bi, 0, 32);
    bv = __shfl(bv, 0, 32);
    if (lane == 0) {
      nbr[(size_t)c * 32 + j] = bi;
      d2[bi] = 3.4e38f;
    }
    if (bv <= r2) mb |= (1u << j);
    __syncthreads();
  }
  if (lane == 0) maskb[c] = mb;
}

// ---------------------------------------------------------------------------
// Gather [x_j, pos_j - pos_i, 0-pad] into f16 activation matrix A [S*32, K0]
// for one graph g.
// ---------------------------------------------------------------------------
__global__ void gather_kernel(const float* __restrict__ feat, const float* __restrict__ pos,
                              const float* __restrict__ posdst, const int* __restrict__ nbr,
                              _Float16* __restrict__ A, int g, int P, int S, int C, int K0) {
  long total = (long)S * 32 * K0;
  for (long e = (long)blockIdx.x * blockDim.x + threadIdx.x; e < total;
       e += (long)gridDim.x * blockDim.x) {
    int col = (int)(e % K0);
    long row = e / K0;
    int s = (int)(row >> 5);
    int j = (int)(row & 31);
    int p = nbr[((size_t)g * S + s) * 32 + j];
    float v = 0.f;
    if (col < C) {
      v = feat[((size_t)g * P + p) * C + col];
    } else if (col < C + 3) {
      int d = col - C;
      v = pos[((size_t)g * P + p) * 3 + d] - posdst[((size_t)g * S + s) * 3 + d];
    }
    A[e] = (_Float16)v;
  }
}

// ---------------------------------------------------------------------------
// H = relu(A @ W + b), f16 in / f32 accumulate (WMMA) / f16 out.
// One wave computes a 32x32 output tile (2x2 WMMA frags).
// A: [M,K] f16 row-major; Wt: [N,K] f16 (pre-transposed); H: [M,N] f16.
// ---------------------------------------------------------------------------
__global__ void __launch_bounds__(128) gemm_relu_kernel(const _Float16* __restrict__ A,
                                                        const _Float16* __restrict__ Wt,
                                                        const float* __restrict__ bias,
                                                        _Float16* __restrict__ H,
                                                        int M, int K, int N) {
  const int wave = blockIdx.x * (blockDim.x >> 5) + (threadIdx.x >> 5);
  const int mt = M >> 5, nt = N >> 5;
  if (wave >= mt * nt) return;            // wave-uniform: EXEC stays all-ones for WMMA
  const int m0 = (wave % mt) * 32;
  const int n0 = (wave / mt) * 32;
  v8f c00 = {}, c01 = {}, c10 = {}, c11 = {};
  for (int kb = 0; kb < K; kb += 32) {
    v16h a0 = load_frag(A, K, m0, kb);
    v16h a1 = load_frag(A, K, m0 + 16, kb);
    v16h b0 = load_frag(Wt, K, n0, kb);
    v16h b1 = load_frag(Wt, K, n0 + 16, kb);
    if (kb + 32 < K)   // gfx1250 global_prefetch of next A K-chunk
      __builtin_prefetch((const void*)(A + (size_t)(m0 + (threadIdx.x & 15)) * K + kb + 32), 0, 1);
    c00 = WMMA_F32_F16(a0, b0, c00);
    c01 = WMMA_F32_F16(a0, b1, c01);
    c10 = WMMA_F32_F16(a1, b0, c10);
    c11 = WMMA_F32_F16(a1, b1, c11);
  }
  const int lane = threadIdx.x & 31;
  const int half = lane >> 4, nn = lane & 15;
  const float bn0 = bias[n0 + nn], bn1 = bias[n0 + 16 + nn];
#pragma unroll
  for (int v = 0; v < 8; ++v) {
    int r = v + half * 8;
    float h;
    h = c00[v] + bn0; h = h > 0.f ? h : 0.f; H[(size_t)(m0 + r) * N + n0 + nn] = (_Float16)h;
    h = c01[v] + bn1; h = h > 0.f ? h : 0.f; H[(size_t)(m0 + r) * N + n0 + 16 + nn] = (_Float16)h;
    h = c10[v] + bn0; h = h > 0.f ? h : 0.f; H[(size_t)(m0 + 16 + r) * N + n0 + nn] = (_Float16)h;
    h = c11[v] + bn1; h = h > 0.f ? h : 0.f; H[(size_t)(m0 + 16 + r) * N + n0 + 16 + nn] = (_Float16)h;
  }
}

// ---------------------------------------------------------------------------
// Fused last layer + masked neighbor max-pool.
// One wave per (center s, 16-col tile): rows s*32..s*32+31 (the 32 neighbors),
// masked by maskb bits, max-reduced across rows -> out[g*S+s, n0..n0+15] f32.
// ---------------------------------------------------------------------------
__global__ void __launch_bounds__(32) gemm_max_kernel(const _Float16* __restrict__ A,
                                                      const _Float16* __restrict__ Wt,
                                                      const float* __restrict__ bias,
                                                      const unsigned* __restrict__ maskb,
                                                      float* __restrict__ out,
                                                      int g, int S, int K, int N) {
  const int s = blockIdx.x;
  const int n0 = blockIdx.y * 16;
  const int m0 = s * 32;
  v8f c0 = {}, c1 = {};
  for (int kb = 0; kb < K; kb += 32) {
    v16h b  = load_frag(Wt, K, n0, kb);
    v16h a0 = load_frag(A, K, m0, kb);
    v16h a1 = load_frag(A, K, m0 + 16, kb);
    c0 = WMMA_F32_F16(a0, b, c0);
    c1 = WMMA_F32_F16(a1, b, c1);
  }
  const unsigned mb = maskb[(size_t)g * S + s];
  const int lane = threadIdx.x & 31;
  const int half = lane >> 4, nn = lane & 15;
  float mx = -3.4e38f;
#pragma unroll
  for (int v = 0; v < 8; ++v) {
    int r = v + half * 8;                 // rows 0..15 (c0) and 16..31 (c1)
    if ((mb >> r) & 1u)        mx = fmaxf(mx, c0[v]);
    if ((mb >> (16 + r)) & 1u) mx = fmaxf(mx, c1[v]);
  }
  float o = __shfl_xor(mx, 16, 32);       // combine the two row-halves per column
  mx = fmaxf(mx, o);
  if (lane < 16) out[((size_t)g * S + s) * N + n0 + nn] = mx + bias[n0 + nn];
}

// ---------------------------------------------------------------------------
// Host orchestration
// ---------------------------------------------------------------------------
extern "C" void kernel_launch(void* const* d_in, const int* in_sizes, int n_in,
                              void* d_out, int out_size, void* d_ws, size_t ws_size,
                              hipStream_t stream) {
  (void)in_sizes; (void)n_in; (void)out_size; (void)ws_size;
  const float* x_in   = (const float*)d_in[0];   // [16*4096, 3]
  const float* pos_in = (const float*)d_in[1];   // [16*4096, 3]
  // params flattened (tree order): sa1..sa4, each (W,b)x3 -> d_in[3 + st*6 + l*2]

  static const int   WID[4][4] = {{6, 32, 32, 64}, {67, 64, 64, 128},
                                  {131, 128, 128, 256}, {259, 256, 256, 512}};
  static const int   KPAD0[4]  = {32, 96, 160, 288};
  static const float RR[4]     = {0.05f, 0.1f, 0.2f, 0.4f};

  unsigned char* ws = (unsigned char*)d_ws;
  size_t off = 0;
  auto carve = [&](size_t b) -> void* {
    void* p = ws + off;
    off += (b + 255) & ~(size_t)255;
    return p;
  };
  float*    posbuf0 = (float*)carve((size_t)16 * 2048 * 3 * sizeof(float));
  float*    posbuf1 = (float*)carve((size_t)16 * 1024 * 3 * sizeof(float));
  float*    featbuf0 = (float*)carve((size_t)16 * 2048 * 64 * sizeof(float));   // 8MB
  float*    featbuf1 = (float*)carve((size_t)16 * 2048 * 64 * sizeof(float));   // 8MB
  int*      idxbuf  = (int*)carve((size_t)16 * 2048 * sizeof(int));
  int*      nbrbuf  = (int*)carve((size_t)16 * 2048 * 32 * sizeof(int));
  unsigned* maskbuf = (unsigned*)carve((size_t)16 * 2048 * sizeof(unsigned));
  _Float16* wtbuf   = (_Float16*)carve((size_t)1 << 20);                        // 1MB
  _Float16* Abuf    = (_Float16*)carve((size_t)1024 * 32 * 96 * sizeof(_Float16));
  _Float16* H1      = (_Float16*)carve((size_t)2048 * 32 * 32 * sizeof(_Float16));
  _Float16* H2      = (_Float16*)carve((size_t)2048 * 32 * 32 * sizeof(_Float16));

  // Convert / transpose / pad all 12 weight matrices to f16 [N, Kpad].
  _Float16* wt[4][3];
  size_t w_off = 0;
  for (int st = 0; st < 4; ++st) {
    for (int l = 0; l < 3; ++l) {
      int K = WID[st][l], N = WID[st][l + 1];
      int Kp = (l == 0) ? KPAD0[st] : K;
      wt[st][l] = wtbuf + w_off;
      w_off += (size_t)N * Kp;
      const float* W = (const float*)d_in[3 + st * 6 + l * 2];
      int total = N * Kp;
      wcvt_kernel<<<(total + 255) / 256, 256, 0, stream>>>(W, wt[st][l], K, N, Kp);
    }
  }

  const float* curpos  = pos_in;
  const float* curfeat = x_in;
  int curC = 3;
  for (int st = 0; st < 4; ++st) {
    const int P = 4096 >> st, S = P >> 1;
    const float r2 = RR[st] * RR[st];
    float* posdst = (st & 1) ? posbuf1 : posbuf0;

    fps_kernel<<<16, FPS_T, (size_t)(P + 2 * FPS_T) * sizeof(float), stream>>>(
        curpos, posdst, idxbuf, P, S);
    nbr_kernel<<<16 * S, 32, (size_t)P * sizeof(float), stream>>>(
        curpos, posdst, nbrbuf, maskbuf, P, S, r2);

    const int K0 = KPAD0[st];
    const int N1 = WID[st][1], N2 = WID[st][2], N3 = WID[st][3];
    float* outfeat = (st == 3) ? (float*)d_out : ((st & 1) ? featbuf1 : featbuf0);
    const float* b1 = (const float*)d_in[3 + st * 6 + 1];
    const float* b2 = (const float*)d_in[3 + st * 6 + 3];
    const float* b3 = (const float*)d_in[3 + st * 6 + 5];
    const int M = S * 32;

    for (int g = 0; g < 16; ++g) {
      long tot = (long)M * K0;
      int gb = (int)((tot + 255) / 256);
      if (gb > 16384) gb = 16384;
      gather_kernel<<<gb, 256, 0, stream>>>(curfeat, curpos, posdst, nbrbuf,
                                            Abuf, g, P, S, curC, K0);
      {
        int waves = (M >> 5) * (N1 >> 5);
        gemm_relu_kernel<<<(waves + 3) / 4, 128, 0, stream>>>(Abuf, wt[st][0], b1, H1, M, K0, N1);
      }
      {
        int waves = (M >> 5) * (N2 >> 5);
        gemm_relu_kernel<<<(waves + 3) / 4, 128, 0, stream>>>(H1, wt[st][1], b2, H2, M, N1, N2);
      }
      gemm_max_kernel<<<dim3(S, N3 / 16), 32, 0, stream>>>(H2, wt[st][2], b3, maskbuf,
                                                           outfeat, g, S, N2, N3);
    }
    curpos = posdst;
    curfeat = outfeat;
    curC = N3;
  }
}
